// TopKSAE_3985729651173
// MI455X (gfx1250) — matrix operand order repaired and checked by reading, not compiled, not run on hardware
//
#include <hip/hip_runtime.h>
#include <hip/hip_bf16.h>
#include <stdint.h>

// ---------------------------------------------------------------------------
// TopK SAE forward on MI455X (gfx1250, wave32, WMMA)
//   encode  : bf16 WMMA GEMM (f32 accum), bias fused
//   topk    : per-row radix select over float bit patterns in LDS
//   decode  : sparse gather-FMA against pre-transposed W_dec (L2-resident)
// ---------------------------------------------------------------------------

typedef __bf16 bf16_t;
typedef __attribute__((ext_vector_type(16))) __bf16 v16bf;
typedef __attribute__((ext_vector_type(8)))  float  v8f;

#define B_ROWS 8192
#define D_IN   768
#define D_LAT  12288
#define K_TOP  64

union FragU {
  v16bf v;
  uint4 u[2];
};

// ---------------- prep: x0 = bf16(x - b_pre) -------------------------------
__global__ void sae_prep_x0(const float* __restrict__ x,
                            const float* __restrict__ b_pre,
                            bf16_t* __restrict__ x0) {
  int i = blockIdx.x * blockDim.x + threadIdx.x;
  if (i < B_ROWS * D_IN) {
    int c = i % D_IN;
    x0[i] = (bf16_t)(x[i] - b_pre[c]);
  }
}

// ---------------- prep: bf16(W_enc) ----------------------------------------
__global__ void sae_prep_wenc(const float* __restrict__ w,
                              bf16_t* __restrict__ wb) {
  int i = blockIdx.x * blockDim.x + threadIdx.x;
  if (i < D_LAT * D_IN) wb[i] = (bf16_t)w[i];
}

// ------------- transpose W_dec (768 x 12288) -> W_decT (12288 x 768) -------
__global__ void sae_transpose_wdec(const float* __restrict__ in,
                                   float* __restrict__ out) {
  __shared__ float tile[32][33];
  int c0 = blockIdx.x * 32;  // latent dim
  int r0 = blockIdx.y * 32;  // input dim
  int tx = threadIdx.x, ty = threadIdx.y;  // block (32, 8)
#pragma unroll
  for (int j = 0; j < 4; ++j)
    tile[ty + 8 * j][tx] = in[(size_t)(r0 + ty + 8 * j) * D_LAT + (c0 + tx)];
  __syncthreads();
#pragma unroll
  for (int j = 0; j < 4; ++j)
    out[(size_t)(c0 + ty + 8 * j) * D_IN + (r0 + tx)] = tile[tx][ty + 8 * j];
}

// ---------------- encode: a = x0 @ W_encT + b_enc  (bf16 WMMA) -------------
// grid = (D_LAT/64, B_ROWS/128), block = 256 (8 waves).
// Wave w computes rows [m0, m0+16), 4 N-tiles of 16 -> 16x64 of `a`.
__global__ void __launch_bounds__(256)
sae_encode_wmma(const bf16_t* __restrict__ x0,
                const bf16_t* __restrict__ wenc,
                const float* __restrict__ b_enc,
                float* __restrict__ a_out) {
  int lane  = threadIdx.x & 31;
  int wave  = threadIdx.x >> 5;
  int lm    = lane & 15;   // M row (A) / N col (B) within tile
  int khalf = lane >> 4;   // which K half this lane holds
  int m0 = blockIdx.y * 128 + wave * 16;
  int n0 = blockIdx.x * 64;

  const bf16_t* arow = x0 + (size_t)(m0 + lm) * D_IN;

  v8f acc[4] = {};

  for (int k0 = 0; k0 < D_IN; k0 += 32) {
    // A 16x32 bf16 fragment (ISA 7.12.2): lane<16 holds K{0..7,16..23},
    // lane>=16 holds K{8..15,24..31}, row M = lane&15.
    FragU A;
    A.u[0] = *(const uint4*)(arow + k0 + khalf * 8);
    A.u[1] = *(const uint4*)(arow + k0 + 16 + khalf * 8);
#pragma unroll
    for (int t = 0; t < 4; ++t) {
      // B 32x16 bf16 fragment: lane = column N, lanes 0-15 hold K{0..7,16..23},
      // lanes 16-31 hold K{8..15,24..31} (same order as A).
      const bf16_t* bptr =
          wenc + (size_t)(n0 + t * 16 + lm) * D_IN + k0 + khalf * 8;
      FragU Bf;
      Bf.u[0] = *(const uint4*)(bptr);
      Bf.u[1] = *(const uint4*)(bptr + 16);
      acc[t] = __builtin_amdgcn_wmma_f32_16x16x32_bf16(
          false, A.v, false, Bf.v, (short)0, acc[t], false, false);
    }
  }

  // C/D layout: VGPR v, lanes 0-15 -> M=v, lanes 16-31 -> M=v+8; N = lane&15.
#pragma unroll
  for (int t = 0; t < 4; ++t) {
    int col = n0 + t * 16 + lm;
    float bias = b_enc[col];
    float* outp = a_out + (size_t)(m0 + 8 * khalf) * D_LAT + col;
#pragma unroll
    for (int v = 0; v < 8; ++v)
      outp[(size_t)v * D_LAT] = acc[t][v] + bias;
  }
}

// ---------------- topk: per-row radix select -------------------------------
// One 256-thread block per row. ReLU'd values staged in LDS; 4-pass MSB-first
// radix select on the (non-negative, order-preserving) float bit patterns.
__global__ void __launch_bounds__(256)
sae_topk(const float* __restrict__ a,
         float* __restrict__ z,
         int* __restrict__ idx_out,
         float* __restrict__ val_out) {
  __shared__ unsigned int vals[D_LAT];   // 48 KB
  __shared__ int counts[256];
  __shared__ unsigned int s_prefix;
  __shared__ int s_kk, s_tie, s_cnt;

  int row = blockIdx.x;
  int tid = threadIdx.x;
  const float* arow = a + (size_t)row * D_LAT;

  for (int i = tid; i < D_LAT; i += 256) {
    float v = arow[i];
    v = v > 0.0f ? v : 0.0f;
    vals[i] = __float_as_uint(v);
  }
  if (tid == 0) { s_tie = 0; s_cnt = 0; }
  __syncthreads();

  unsigned int prefix = 0;
  int kk = K_TOP;
#pragma unroll
  for (int p = 3; p >= 0; --p) {
    counts[tid] = 0;
    __syncthreads();
    unsigned int pmask = (p == 3) ? 0u : (0xFFFFFFFFu << ((p + 1) * 8));
    for (int i = tid; i < D_LAT; i += 256) {
      unsigned int b = vals[i];
      if ((b & pmask) == prefix)
        atomicAdd(&counts[(b >> (p * 8)) & 0xFF], 1);
    }
    __syncthreads();
    if (tid == 0) {
      int above = 0, sel = 0;
      for (int b = 255; b >= 0; --b) {
        int c = counts[b];
        if (above + c >= kk) { sel = b; break; }
        above += c;
      }
      s_prefix = prefix | ((unsigned int)sel << (p * 8));
      s_kk = kk - above;
    }
    __syncthreads();
    prefix = s_prefix;
    kk = s_kk;
  }

  unsigned int Tbits = prefix;   // k-th largest value (bit pattern)
  int ties_needed = kk;          // how many == T to admit
  float* zrow = z + (size_t)row * D_LAT;

  for (int i = tid; i < D_LAT; i += 256) {
    unsigned int b = vals[i];
    float v = __uint_as_float(b);
    bool admit = false;
    if (b > Tbits) {
      admit = true;
    } else if (b == Tbits) {
      int slot = atomicAdd(&s_tie, 1);
      if (slot < ties_needed) admit = true;
    }
    __builtin_nontemporal_store(admit ? v : 0.0f, &zrow[i]);  // z: write-once
    if (admit) {
      int pos = atomicAdd(&s_cnt, 1);       // exactly K_TOP total per row
      idx_out[(size_t)row * K_TOP + pos] = i;
      val_out[(size_t)row * K_TOP + pos] = v;
    }
  }
}

// ---------------- decode: x_hat = b_dec + sum val_j * W_decT[idx_j, :] -----
__global__ void __launch_bounds__(256)
sae_decode(const float* __restrict__ wdecT,
           const float* __restrict__ b_dec,
           const int* __restrict__ idx_in,
           const float* __restrict__ val_in,
           float* __restrict__ xhat) {
  __shared__ int   sidx[K_TOP];
  __shared__ float sval[K_TOP];
  int row = blockIdx.x;
  int tid = threadIdx.x;
  if (tid < K_TOP) {
    sidx[tid] = idx_in[(size_t)row * K_TOP + tid];
    sval[tid] = val_in[(size_t)row * K_TOP + tid];
  }
  __syncthreads();
  for (int i = tid; i < D_IN; i += 256) {
    float acc = b_dec[i];
#pragma unroll 8
    for (int j = 0; j < K_TOP; ++j)
      acc += sval[j] * wdecT[(size_t)sidx[j] * D_IN + i];
    xhat[(size_t)row * D_IN + i] = acc;
  }
}

// ---------------------------------------------------------------------------
extern "C" void kernel_launch(void* const* d_in, const int* in_sizes, int n_in,
                              void* d_out, int out_size, void* d_ws, size_t ws_size,
                              hipStream_t stream) {
  const float* x     = (const float*)d_in[0];
  const float* b_pre = (const float*)d_in[1];
  const float* W_enc = (const float*)d_in[2];
  const float* b_enc = (const float*)d_in[3];
  const float* W_dec = (const float*)d_in[4];
  const float* b_dec = (const float*)d_in[5];
  // d_in[6] = k (scalar, ==64, fixed at compile time)

  // Workspace carve-up (all offsets 256B aligned).
  uint8_t* ws = (uint8_t*)d_ws;
  size_t off = 0;
  bf16_t* x0    = (bf16_t*)(ws + off); off += (size_t)B_ROWS * D_IN * sizeof(bf16_t); // 12.6 MB
  bf16_t* wencb = (bf16_t*)(ws + off); off += (size_t)D_LAT * D_IN * sizeof(bf16_t);  // 18.9 MB
  float*  wdecT = (float*)(ws + off);  off += (size_t)D_LAT * D_IN * sizeof(float);   // 37.7 MB
  int*    idxb  = (int*)(ws + off);    off += (size_t)B_ROWS * K_TOP * sizeof(int);   //  2.1 MB
  float*  valb  = (float*)(ws + off);  off += (size_t)B_ROWS * K_TOP * sizeof(float); //  2.1 MB

  // Outputs: (x_hat, z, a) concatenated flat, f32.
  float* out  = (float*)d_out;
  float* xhat = out;
  float* zbuf = out + (size_t)B_ROWS * D_IN;
  float* abuf = zbuf + (size_t)B_ROWS * D_LAT;

  sae_prep_x0<<<(B_ROWS * D_IN + 255) / 256, 256, 0, stream>>>(x, b_pre, x0);
  sae_prep_wenc<<<(D_LAT * D_IN + 255) / 256, 256, 0, stream>>>(W_enc, wencb);
  sae_transpose_wdec<<<dim3(D_LAT / 32, D_IN / 32), dim3(32, 8), 0, stream>>>(W_dec, wdecT);
  sae_encode_wmma<<<dim3(D_LAT / 64, B_ROWS / 128), 256, 0, stream>>>(x0, wencb, b_enc, abuf);
  sae_topk<<<B_ROWS, 256, 0, stream>>>(abuf, zbuf, idxb, valb);
  sae_decode<<<B_ROWS, 256, 0, stream>>>(wdecT, b_dec, idxb, valb, xhat);
}
